// Transformer_59373627900238
// MI455X (gfx1250) — matrix operand order repaired and checked by reading, not compiled, not run on hardware
//
#include <hip/hip_runtime.h>
#include <hip/hip_bf16.h>
#include <math.h>

typedef __attribute__((ext_vector_type(16))) _Float16 v16h;
typedef __attribute__((ext_vector_type(8)))  _Float16 v8h;
typedef __attribute__((ext_vector_type(8)))  float    v8f;

#define D_MODEL 384
#define NHEAD   12
#define HDIM    32

// ---------------------------------------------------------------------------
// prep: x = src + positional_encoding (f32 + f16 shadow),
//       mask[token] = any(src[token,:] != 0). One wave per token.
// ---------------------------------------------------------------------------
__global__ __launch_bounds__(32)
void prep_kernel(const float* __restrict__ src, float* __restrict__ dst,
                 _Float16* __restrict__ dsth, float* __restrict__ mask, int Ltok)
{
    int tok  = blockIdx.x;            // 0 .. B*Ltok-1
    int pos  = tok % Ltok;
    int lane = threadIdx.x;
    const float* s = src  + (size_t)tok * D_MODEL;
    float*       d = dst  + (size_t)tok * D_MODEL;
    _Float16*    dh = dsth + (size_t)tok * D_MODEL;
    float nz = 0.f;
    #pragma unroll
    for (int j = 0; j < D_MODEL / 32; ++j) {
        int i = lane + 32 * j;
        float v = s[i];
        nz += (v != 0.f) ? 1.f : 0.f;
        int   ip   = i & ~1;                       // pair-shared frequency
        float freq = __powf(10000.f, (float)ip / (float)D_MODEL);
        float ang  = (float)pos / freq;
        float pe   = (i & 1) ? __cosf(ang) : __sinf(ang);
        float o    = v + pe;
        d[i]  = o;
        dh[i] = (_Float16)o;
    }
    #pragma unroll
    for (int off = 16; off; off >>= 1) nz += __shfl_xor(nz, off, 32);
    if (lane == 0) mask[tok] = (nz > 0.f) ? 1.f : 0.f;
}

// ---------------------------------------------------------------------------
// weight transpose + f32 -> f16:  Wt[n*K + k] = (half)W[k*N + n]
// ---------------------------------------------------------------------------
__global__ void wt_kernel(const float* __restrict__ W, _Float16* __restrict__ Wt,
                          int K, int N)
{
    int total = K * N;
    for (int idx = blockIdx.x * blockDim.x + threadIdx.x; idx < total;
         idx += gridDim.x * blockDim.x) {
        int k = idx / N, n = idx % N;
        Wt[(size_t)n * K + k] = (_Float16)W[idx];
    }
}

// ---------------------------------------------------------------------------
// residual add + LayerNorm (in place on x, plus f16 shadow). One wave per row.
// ---------------------------------------------------------------------------
__global__ __launch_bounds__(32)
void add_ln_kernel(float* __restrict__ x, _Float16* __restrict__ xh,
                   const float* __restrict__ a,
                   const float* __restrict__ gamma, const float* __restrict__ beta)
{
    int row  = blockIdx.x;
    int lane = threadIdx.x;
    float*       xr  = x  + (size_t)row * D_MODEL;
    _Float16*    xhr = xh + (size_t)row * D_MODEL;
    const float* ar  = a  + (size_t)row * D_MODEL;
    float v[D_MODEL / 32];
    float s = 0.f;
    #pragma unroll
    for (int j = 0; j < D_MODEL / 32; ++j) {
        v[j] = xr[lane + 32 * j] + ar[lane + 32 * j];
        s += v[j];
    }
    #pragma unroll
    for (int off = 16; off; off >>= 1) s += __shfl_xor(s, off, 32);
    float mean = s / (float)D_MODEL;
    float vs = 0.f;
    #pragma unroll
    for (int j = 0; j < D_MODEL / 32; ++j) { float dd = v[j] - mean; vs += dd * dd; }
    #pragma unroll
    for (int off = 16; off; off >>= 1) vs += __shfl_xor(vs, off, 32);
    float rstd = rsqrtf(vs / (float)D_MODEL + 1e-6f);
    #pragma unroll
    for (int j = 0; j < D_MODEL / 32; ++j) {
        int i = lane + 32 * j;
        float o = gamma[i] * (v[j] - mean) * rstd + beta[i];
        xr[i]  = o;
        xhr[i] = (_Float16)o;
    }
}

// ---------------------------------------------------------------------------
// GEMM: C[M,N] = Ah[M,K](f16) * Wt^T(f16, stored [N][K]) + bias.
// One wave computes a 16x64 tile: A fragment reused across 4 WMMAs per K-step.
// mode 0: write f32 C (optional relu); mode 1: write f16 Ch (optional relu).
// ---------------------------------------------------------------------------
__global__ __launch_bounds__(32)
void gemm_kernel(const _Float16* __restrict__ Ah, const _Float16* __restrict__ Wt,
                 const float* __restrict__ bias,
                 float* __restrict__ C, _Float16* __restrict__ Ch,
                 int M, int K, int N, int relu, int mode)
{
    int nt = blockIdx.x, mt = blockIdx.y;       // nt: 64-wide column block
    int lane = threadIdx.x & 31;
    int half = lane >> 4;
    int l15  = lane & 15;

    int arow = mt * 16 + l15; if (arow >= M) arow = M - 1;
    const _Float16* Abase = Ah + (size_t)arow * K;
    const _Float16* Wbase[4];
    #pragma unroll
    for (int i = 0; i < 4; ++i)
        Wbase[i] = Wt + (size_t)(nt * 64 + i * 16 + l15) * K;

    v8f acc0 = {}, acc1 = {}, acc2 = {}, acc3 = {};
    for (int kb = 0; kb < K; kb += 32) {
        __builtin_prefetch((const void*)(Abase + kb + 128), 0, 0);
        // A fragment: two contiguous 16B chunks (K = half*16+0..7, half*16+8..15)
        v8h alo = *(const v8h*)(Abase + kb + half * 16);
        v8h ahi = *(const v8h*)(Abase + kb + half * 16 + 8);
        v16h af = __builtin_shufflevector(alo, ahi, 0,1,2,3,4,5,6,7,
                                                    8,9,10,11,12,13,14,15);
        v16h b0 = *(const v16h*)(Wbase[0] + kb + half * 16);
        v16h b1 = *(const v16h*)(Wbase[1] + kb + half * 16);
        v16h b2 = *(const v16h*)(Wbase[2] + kb + half * 16);
        v16h b3 = *(const v16h*)(Wbase[3] + kb + half * 16);
        acc0 = __builtin_amdgcn_wmma_f32_16x16x32_f16(false, af, false, b0, (short)0, acc0, false, false);
        acc1 = __builtin_amdgcn_wmma_f32_16x16x32_f16(false, af, false, b1, (short)0, acc1, false, false);
        acc2 = __builtin_amdgcn_wmma_f32_16x16x32_f16(false, af, false, b2, (short)0, acc2, false, false);
        acc3 = __builtin_amdgcn_wmma_f32_16x16x32_f16(false, af, false, b3, (short)0, acc3, false, false);
    }
    v8f* accs[4] = {&acc0, &acc1, &acc2, &acc3};
    #pragma unroll
    for (int i = 0; i < 4; ++i) {
        int col = nt * 64 + i * 16 + l15;
        float bv = bias[col];
        #pragma unroll
        for (int j = 0; j < 8; ++j) {
            int row = mt * 16 + half * 8 + j;
            if (row < M) {
                float cv = (*accs[i])[j] + bv;
                if (relu) cv = fmaxf(cv, 0.f);
                if (mode == 0) C[(size_t)row * N + col] = cv;
                else           Ch[(size_t)row * N + col] = (_Float16)cv;
            }
        }
    }
}

// ---------------------------------------------------------------------------
// Attention (Q = x, K = V = kv, per-head scalar scale, value mask -> -1e9,
// query mask zeroes output rows). Grid: (ceil(Lq/16), NHEAD, B), one wave.
// Dynamic LDS: f32 scores [16][Lk_pad] + rinv[16] + f16 probs [16][Lk_pad].
// ---------------------------------------------------------------------------
__global__ __launch_bounds__(32)
void attn_kernel(const _Float16* __restrict__ Qh, const _Float16* __restrict__ KVh,
                 const float* __restrict__ qmask, const float* __restrict__ kmask,
                 const float* __restrict__ scales, float* __restrict__ Out,
                 int Lq, int Lk, int Lk_pad)
{
    extern __shared__ float smem[];
    float*    sS = smem;                                  // [16][Lk_pad] f32
    float*    sR = smem + 16 * Lk_pad;                    // [16]
    _Float16* sP = (_Float16*)(smem + 16 * Lk_pad + 16);  // [16][Lk_pad] f16

    int qt = blockIdx.x, h = blockIdx.y, bz = blockIdx.z;
    int lane = threadIdx.x & 31;
    int half = lane >> 4, l15 = lane & 15;
    float scale = scales[h];

    int qrow = qt * 16 + l15; if (qrow >= Lq) qrow = Lq - 1;
    const _Float16* qbase = Qh + ((size_t)(bz * Lq + qrow) * D_MODEL + h * HDIM);

    // A fragment (16x32 Q tile over head dim) -- reused for every key tile
    v8h qlo = *(const v8h*)(qbase + half * 16);
    v8h qhi = *(const v8h*)(qbase + half * 16 + 8);
    v16h aq = __builtin_shufflevector(qlo, qhi, 0,1,2,3,4,5,6,7,
                                                8,9,10,11,12,13,14,15);

    // ---- pass 1: scores = scale * Q K^T, masked, into LDS ----
    for (int kt = 0; kt < Lk_pad / 16; ++kt) {
        int kcol = kt * 16 + l15;
        int kc   = (kcol < Lk) ? kcol : (Lk - 1);
        v16h bf = *(const v16h*)(KVh + ((size_t)(bz * Lk + kc) * D_MODEL
                                        + h * HDIM + half * 16));
        v8f c = {};
        c = __builtin_amdgcn_wmma_f32_16x16x32_f16(false, aq, false, bf,
                                                   (short)0, c, false, false);
        bool valid = (kcol < Lk) && (kmask[bz * Lk + kc] != 0.f);
        #pragma unroll
        for (int j = 0; j < 8; ++j) {
            int m = half * 8 + j;
            sS[m * Lk_pad + kcol] = valid ? c[j] * scale : -1e9f;
        }
    }
    __syncthreads();

    // ---- pass 2: row softmax; unnormalized exp -> f16 probs, keep 1/sum ----
    for (int r = 0; r < 16; ++r) {
        const float* row = sS + r * Lk_pad;
        _Float16*    prow = sP + r * Lk_pad;
        float mx = -3.4e38f;
        for (int c = lane; c < Lk_pad; c += 32) mx = fmaxf(mx, row[c]);
        #pragma unroll
        for (int off = 16; off; off >>= 1) mx = fmaxf(mx, __shfl_xor(mx, off, 32));
        float sum = 0.f;
        for (int c = lane; c < Lk_pad; c += 32) {
            float e = __expf(row[c] - mx);
            prow[c] = (_Float16)e;
            sum += e;
        }
        #pragma unroll
        for (int off = 16; off; off >>= 1) sum += __shfl_xor(sum, off, 32);
        if (lane == 0) sR[r] = 1.f / sum;
    }
    __syncthreads();

    // ---- pass 3: out = P @ V (two 16-col halves of the 32-wide head) ----
    v8f acc0 = {}, acc1 = {};
    for (int kb = 0; kb < Lk_pad; kb += 32) {
        const _Float16* prow = sP + (size_t)l15 * Lk_pad + kb;
        v8h plo = *(const v8h*)(prow + half * 16);
        v8h phi = *(const v8h*)(prow + half * 16 + 8);
        v16h af = __builtin_shufflevector(plo, phi, 0,1,2,3,4,5,6,7,
                                                    8,9,10,11,12,13,14,15);
        v16h b0, b1;
        #pragma unroll
        for (int t = 0; t < 16; ++t) {
            int key = kb + half * 16 + t; if (key >= Lk) key = Lk - 1;
            const _Float16* vb = KVh + ((size_t)(bz * Lk + key) * D_MODEL + h * HDIM);
            b0[t] = vb[l15];
            b1[t] = vb[16 + l15];
        }
        acc0 = __builtin_amdgcn_wmma_f32_16x16x32_f16(false, af, false, b0,
                                                      (short)0, acc0, false, false);
        acc1 = __builtin_amdgcn_wmma_f32_16x16x32_f16(false, af, false, b1,
                                                      (short)0, acc1, false, false);
    }
    #pragma unroll
    for (int j = 0; j < 8; ++j) {
        int m = half * 8 + j;
        int row = qt * 16 + m;
        if (row < Lq) {
            float f = sR[m] * qmask[bz * Lq + row];
            float* o = Out + ((size_t)(bz * Lq + row) * D_MODEL + h * HDIM);
            o[l15]      = acc0[j] * f;
            o[16 + l15] = acc1[j] * f;
        }
    }
}

// ---------------------------------------------------------------------------
// host-side orchestration
// ---------------------------------------------------------------------------
extern "C" void kernel_launch(void* const* d_in, const int* in_sizes, int n_in,
                              void* d_out, int out_size, void* d_ws, size_t ws_size,
                              hipStream_t stream)
{
    (void)in_sizes; (void)n_in; (void)out_size; (void)ws_size;
    const float* enc_in    = (const float*)d_in[0];
    const float* dec_in    = (const float*)d_in[1];
    const float* enc_scale = (const float*)d_in[2];
    const float* enc_ln    = (const float*)d_in[3];
    const float* enc_w1    = (const float*)d_in[4];
    const float* enc_b1    = (const float*)d_in[5];
    const float* enc_w2    = (const float*)d_in[6];
    const float* enc_b2    = (const float*)d_in[7];
    const float* dec_scale = (const float*)d_in[8];
    const float* dec_ln    = (const float*)d_in[9];
    const float* dec_w1    = (const float*)d_in[10];
    const float* dec_b1    = (const float*)d_in[11];
    const float* dec_w2    = (const float*)d_in[12];
    const float* dec_b2    = (const float*)d_in[13];
    const float* final_w   = (const float*)d_in[14];
    const float* final_b   = (const float*)d_in[15];

    const int B = 2, Le = 1024, Ld = 1023, D = D_MODEL, F = 1536, H = NHEAD;
    const int Me = B * Le;          // 2048
    const int Md = B * Ld;          // 2046
    const int Lkp = 1024;           // padded key length (mult of 32)

    char* ws = (char*)d_ws;
    size_t off = 0;
    auto alloc = [&](size_t bytes) -> char* {
        char* p = ws + off;
        off += (bytes + 255) & ~(size_t)255;
        return p;
    };
    float*    x_enc  = (float*)alloc((size_t)Me * D * 4);
    float*    y_dec  = (float*)alloc((size_t)Me * D * 4);
    float*    abuf   = (float*)alloc((size_t)Me * D * 4);
    _Float16* xh_enc = (_Float16*)alloc((size_t)Me * D * 2);
    _Float16* yh_dec = (_Float16*)alloc((size_t)Me * D * 2);
    _Float16* t1h    = (_Float16*)alloc((size_t)Me * F * 2);
    float*    mask_e = (float*)alloc((size_t)B * Le * 4);
    float*    mask_d = (float*)alloc((size_t)B * Ld * 4);
    _Float16* ew1t[2], *ew2t[2], *dw1t[2], *dw2t[2];
    for (int l = 0; l < 2; ++l) ew1t[l] = (_Float16*)alloc((size_t)D * F * 2);
    for (int l = 0; l < 2; ++l) ew2t[l] = (_Float16*)alloc((size_t)F * D * 2);
    for (int l = 0; l < 2; ++l) dw1t[l] = (_Float16*)alloc((size_t)D * F * 2);
    for (int l = 0; l < 2; ++l) dw2t[l] = (_Float16*)alloc((size_t)F * D * 2);
    _Float16* fwt = (_Float16*)alloc((size_t)D * D * 2);

    // 1) positional encoding + masks (+ f16 shadows)
    prep_kernel<<<B * Le, 32, 0, stream>>>(enc_in, x_enc, xh_enc, mask_e, Le);
    prep_kernel<<<B * Ld, 32, 0, stream>>>(dec_in, y_dec, yh_dec, mask_d, Ld);

    // 2) weights -> transposed f16
    int blkDF = (D * F + 255) / 256;
    for (int l = 0; l < 2; ++l) {
        wt_kernel<<<blkDF, 256, 0, stream>>>(enc_w1 + (size_t)l * D * F, ew1t[l], D, F);
        wt_kernel<<<blkDF, 256, 0, stream>>>(enc_w2 + (size_t)l * F * D, ew2t[l], F, D);
        wt_kernel<<<blkDF, 256, 0, stream>>>(dec_w1 + (size_t)l * D * F, dw1t[l], D, F);
        wt_kernel<<<blkDF, 256, 0, stream>>>(dec_w2 + (size_t)l * F * D, dw2t[l], F, D);
    }
    wt_kernel<<<(D * D + 255) / 256, 256, 0, stream>>>(final_w, fwt, D, D);

    // dynamic LDS: f32 scores + rinv + f16 probs
    size_t shA = (size_t)(16 * Lkp + 16) * 4 + (size_t)16 * Lkp * 2;

    // 3) encoder
    for (int l = 0; l < 2; ++l) {
        attn_kernel<<<dim3((Le + 15) / 16, H, B), 32, shA, stream>>>(
            xh_enc, xh_enc, mask_e, mask_e, enc_scale + l * H, abuf, Le, Le, Lkp);
        add_ln_kernel<<<Me, 32, 0, stream>>>(
            x_enc, xh_enc, abuf,
            enc_ln + ((size_t)(l * 2 + 0) * 2 + 0) * D,
            enc_ln + ((size_t)(l * 2 + 0) * 2 + 1) * D);
        gemm_kernel<<<dim3(F / 64, (Me + 15) / 16), 32, 0, stream>>>(
            xh_enc, ew1t[l], enc_b1 + l * F, nullptr, t1h, Me, D, F, 1, 1);
        gemm_kernel<<<dim3(D / 64, (Me + 15) / 16), 32, 0, stream>>>(
            t1h, ew2t[l], enc_b2 + l * D, abuf, nullptr, Me, F, D, 0, 0);
        add_ln_kernel<<<Me, 32, 0, stream>>>(
            x_enc, xh_enc, abuf,
            enc_ln + ((size_t)(l * 2 + 1) * 2 + 0) * D,
            enc_ln + ((size_t)(l * 2 + 1) * 2 + 1) * D);
    }

    // 4) decoder
    for (int l = 0; l < 2; ++l) {
        // self attention (no causal mask in original)
        attn_kernel<<<dim3((Ld + 15) / 16, H, B), 32, shA, stream>>>(
            yh_dec, yh_dec, mask_d, mask_d, dec_scale + (l * 2 + 0) * H, abuf, Ld, Ld, Lkp);
        add_ln_kernel<<<Md, 32, 0, stream>>>(
            y_dec, yh_dec, abuf,
            dec_ln + ((size_t)(l * 3 + 0) * 2 + 0) * D,
            dec_ln + ((size_t)(l * 3 + 0) * 2 + 1) * D);
        // cross attention with encoder output
        attn_kernel<<<dim3((Ld + 15) / 16, H, B), 32, shA, stream>>>(
            yh_dec, xh_enc, mask_d, mask_e, dec_scale + (l * 2 + 1) * H, abuf, Ld, Le, Lkp);
        add_ln_kernel<<<Md, 32, 0, stream>>>(
            y_dec, yh_dec, abuf,
            dec_ln + ((size_t)(l * 3 + 1) * 2 + 0) * D,
            dec_ln + ((size_t)(l * 3 + 1) * 2 + 1) * D);
        // FFN
        gemm_kernel<<<dim3(F / 64, (Md + 15) / 16), 32, 0, stream>>>(
            yh_dec, dw1t[l], dec_b1 + l * F, nullptr, t1h, Md, D, F, 1, 1);
        gemm_kernel<<<dim3(D / 64, (Md + 15) / 16), 32, 0, stream>>>(
            t1h, dw2t[l], dec_b2 + l * D, abuf, nullptr, Md, F, D, 0, 0);
        add_ln_kernel<<<Md, 32, 0, stream>>>(
            y_dec, yh_dec, abuf,
            dec_ln + ((size_t)(l * 3 + 2) * 2 + 0) * D,
            dec_ln + ((size_t)(l * 3 + 2) * 2 + 1) * D);
    }

    // 5) final projection -> d_out (f32, [B*Ld, D])
    gemm_kernel<<<dim3(D / 64, (Md + 15) / 16), 32, 0, stream>>>(
        yh_dec, fwt, final_b, (float*)d_out, nullptr, Md, D, D, 0, 0);
}
